// MultiHeadAttention_71236327571522
// MI455X (gfx1250) — compile-verified, matrix operand order checked by reading, not run on hardware
//
#include <hip/hip_runtime.h>
#include <hip/hip_bf16.h>

// MultiHeadAttention for MI455X (gfx1250, wave32, WMMA).
// Phases: f32->f16 convert, QKV GEMM (scale folded into Q, V stored transposed),
// flash attention with online softmax (causal + key-padding mask), output GEMM.
// All matmuls on v_wmma_f32_16x16x32_f16 with fp32 accumulation.
// v2: block-uniform Q/K/V epilogue split (scalar branch instead of per-lane
//     exec predication), global_prefetch in GEMM k-loops.

typedef __attribute__((ext_vector_type(16))) _Float16 v16h;
typedef __attribute__((ext_vector_type(8)))  _Float16 v8h;
typedef __attribute__((ext_vector_type(4)))  _Float16 v4h;
typedef __attribute__((ext_vector_type(8)))  float    v8f;

#define B_   4
#define S_   2048
#define DIM_ 1024
#define NH_  16
#define HD_  64
#define BS_  (B_ * S_)        // 8192 rows
#define E3_  (3 * DIM_)       // 3072 qkv channels

__device__ __forceinline__ v8f wmma16(v16h a, v16h b, v8f c) {
  // v_wmma_f32_16x16x32_f16: (neg_a, A, neg_b, B, c_mod, C, reuse_a, reuse_b)
  return __builtin_amdgcn_wmma_f32_16x16x32_f16(false, a, false, b, (short)0, c,
                                                false, false);
}

// Load a 16x32 f16 A/B fragment from a row-major [rows][ld] array at (row0,k0)=p.
// CDNA5 layout: lane l holds row (l&15); k = 8*(l>>4) + {0..7} and + {16..23}.
// Two 16-byte loads per lane; requires ld % 8 == 0 and 16B-aligned base.
__device__ __forceinline__ v16h load_frag(const _Float16* p, int ld, int lane) {
  const _Float16* q = p + (size_t)(lane & 15) * (size_t)ld + ((lane >> 4) << 3);
  v8h lo = *(const v8h*)q;
  v8h hi = *(const v8h*)(q + 16);
  v16h r;
#pragma unroll
  for (int i = 0; i < 8; ++i) { r[i] = lo[i]; r[i + 8] = hi[i]; }
  return r;
}

// ---------------------------------------------------------------- f32 -> f16
__global__ void cvt_f32_f16(const float* __restrict__ src,
                            _Float16* __restrict__ dst, int n) {
  int idx = (blockIdx.x * blockDim.x + threadIdx.x) * 4;
  if (idx + 3 < n) {
    const float4 v = *(const float4*)(src + idx);
    v4h h;
    h[0] = (_Float16)v.x; h[1] = (_Float16)v.y;
    h[2] = (_Float16)v.z; h[3] = (_Float16)v.w;
    *(v4h*)(dst + idx) = h;
  }
}

// ------------------------------------------------- QKV = X @ Wqkv^T (f16 in/out)
// grid (BS_/128, E3_/128), 256 threads = 8 waves as 2(M) x 4(N); wave tile 64x32.
// The 128-wide N tile sits inside one 1024-wide Q/K/V section, so the
// destination select (c) is block-uniform: c = blockIdx.y >> 3.
__global__ __launch_bounds__(256) void qkv_gemm(const _Float16* __restrict__ x16,
                                                const _Float16* __restrict__ w16,
                                                _Float16* __restrict__ Qf,
                                                _Float16* __restrict__ Kf,
                                                _Float16* __restrict__ Vt) {
  const int lane = threadIdx.x & 31, wave = threadIdx.x >> 5;
  const int m0 = blockIdx.x * 128 + (wave & 1) * 64;
  const int n0 = blockIdx.y * 128 + (wave >> 1) * 32;
  v8f acc[4][2] = {};
  for (int k0 = 0; k0 < DIM_; k0 += 32) {
    __builtin_prefetch(x16 + (size_t)m0 * DIM_ + k0 + 64, 0, 1);
    __builtin_prefetch(w16 + (size_t)n0 * DIM_ + k0 + 64, 0, 1);
    v16h a[4], b[2];
#pragma unroll
    for (int im = 0; im < 4; ++im)
      a[im] = load_frag(x16 + (size_t)(m0 + 16 * im) * DIM_ + k0, DIM_, lane);
#pragma unroll
    for (int in = 0; in < 2; ++in)
      b[in] = load_frag(w16 + (size_t)(n0 + 16 * in) * DIM_ + k0, DIM_, lane);
#pragma unroll
    for (int im = 0; im < 4; ++im)
#pragma unroll
      for (int in = 0; in < 2; ++in)
        acc[im][in] = wmma16(a[im], b[in], acc[im][in]);
  }
  const int half = lane >> 4, col = lane & 15;
  const int c  = blockIdx.y >> 3;                              // 0=Q 1=K 2=V
  const int e0 = (blockIdx.y & 7) * 128 + (wave >> 1) * 32;    // col in section
  if (c == 0) {                                 // Q: [b,h,s,d], fold 1/sqrt(64)
#pragma unroll
    for (int im = 0; im < 4; ++im)
#pragma unroll
      for (int in = 0; in < 2; ++in)
#pragma unroll
        for (int r = 0; r < 8; ++r) {
          const int m = m0 + 16 * im + r + 8 * half;
          const int el = e0 + 16 * in + col;
          const int bb = m >> 11, ss = m & (S_ - 1);
          const int hh = el >> 6, dd = el & (HD_ - 1);
          Qf[(((size_t)(bb * NH_ + hh)) * S_ + ss) * HD_ + dd] =
              (_Float16)(acc[im][in][r] * 0.125f);
        }
  } else if (c == 1) {                          // K: [b,h,s,d]
#pragma unroll
    for (int im = 0; im < 4; ++im)
#pragma unroll
      for (int in = 0; in < 2; ++in)
#pragma unroll
        for (int r = 0; r < 8; ++r) {
          const int m = m0 + 16 * im + r + 8 * half;
          const int el = e0 + 16 * in + col;
          const int bb = m >> 11, ss = m & (S_ - 1);
          const int hh = el >> 6, dd = el & (HD_ - 1);
          Kf[(((size_t)(bb * NH_ + hh)) * S_ + ss) * HD_ + dd] =
              (_Float16)acc[im][in][r];
        }
  } else {                                      // V transposed: [b,h,d,s]
#pragma unroll
    for (int im = 0; im < 4; ++im)
#pragma unroll
      for (int in = 0; in < 2; ++in)
#pragma unroll
        for (int r = 0; r < 8; ++r) {
          const int m = m0 + 16 * im + r + 8 * half;
          const int el = e0 + 16 * in + col;
          const int bb = m >> 11, ss = m & (S_ - 1);
          const int hh = el >> 6, dd = el & (HD_ - 1);
          Vt[(((size_t)(bb * NH_ + hh)) * HD_ + dd) * S_ + ss] =
              (_Float16)acc[im][in][r];
        }
  }
}

// --------------------------------------------------------- flash attention
// grid (S_/128, B_*NH_), 256 threads = 8 independent waves; wave = 16 q-rows.
__global__ __launch_bounds__(256) void attn_kernel(const _Float16* __restrict__ Qf,
                                                   const _Float16* __restrict__ Kf,
                                                   const _Float16* __restrict__ Vt,
                                                   const unsigned char* __restrict__ mask,
                                                   _Float16* __restrict__ O16) {
  __shared__ _Float16 Pl[8][16 * 32];                      // per-wave P tile
  const int lane = threadIdx.x & 31, wave = threadIdx.x >> 5;
  const int half = lane >> 4, col = lane & 15;
  const int bh = blockIdx.y, bb = bh >> 4, hh = bh & (NH_ - 1);
  const int q0 = blockIdx.x * 128 + wave * 16;
  const _Float16* Qp = Qf + (size_t)bh * S_ * HD_;
  const _Float16* Kp = Kf + (size_t)bh * S_ * HD_;
  const _Float16* Vp = Vt + (size_t)bh * HD_ * S_;
  const unsigned char* mp = mask + (size_t)bb * S_;
  _Float16* Pw = &Pl[wave][0];

  const v16h qf0 = load_frag(Qp + (size_t)q0 * HD_, HD_, lane);        // d 0..31
  const v16h qf1 = load_frag(Qp + (size_t)q0 * HD_ + 32, HD_, lane);   // d 32..63

  v8f o[4] = {};
  float rm[8], rl[8];
#pragma unroll
  for (int r = 0; r < 8; ++r) { rm[r] = -1e30f; rl[r] = 0.f; }

  const int kend = q0 + 16;                  // causal: keys <= q0+15 needed
  for (int k0 = 0; k0 < kend; k0 += 32) {
    // ---- S = Q K^T for 32 keys (two 16x16 tiles, chained over d)
    v8f s0 = {}, s1 = {};
    s0 = wmma16(qf0, load_frag(Kp + (size_t)k0 * HD_, HD_, lane), s0);
    s0 = wmma16(qf1, load_frag(Kp + (size_t)k0 * HD_ + 32, HD_, lane), s0);
    s1 = wmma16(qf0, load_frag(Kp + (size_t)(k0 + 16) * HD_, HD_, lane), s1);
    s1 = wmma16(qf1, load_frag(Kp + (size_t)(k0 + 16) * HD_ + 32, HD_, lane), s1);

    const unsigned char mk0 = mp[k0 + col];
    const unsigned char mk1 = mp[k0 + 16 + col];
    // ---- online softmax; lane (half,col) holds rows r+8*half, key col
#pragma unroll
    for (int r = 0; r < 8; ++r) {
      const int qq = q0 + r + 8 * half;
      float v0 = s0[r], v1 = s1[r];
      if (mk0 || (k0 + col) > qq)       v0 = -1e30f;
      if (mk1 || (k0 + 16 + col) > qq)  v1 = -1e30f;
      float mt = fmaxf(v0, v1);
#pragma unroll
      for (int off = 1; off < 16; off <<= 1) mt = fmaxf(mt, __shfl_xor(mt, off));
      const float mnew = fmaxf(rm[r], mt);
      const float scl = __expf(rm[r] - mnew);
      const float p0 = (v0 < -1e29f) ? 0.f : __expf(v0 - mnew);
      const float p1 = (v1 < -1e29f) ? 0.f : __expf(v1 - mnew);
      float rs = p0 + p1;
#pragma unroll
      for (int off = 1; off < 16; off <<= 1) rs += __shfl_xor(rs, off);
      rl[r] = rl[r] * scl + rs;
      rm[r] = mnew;
#pragma unroll
      for (int t = 0; t < 4; ++t) o[t][r] *= scl;
      // stage P (C-layout -> row-major LDS); same-wave DS ops stay in order
      Pw[(r + 8 * half) * 32 + col]      = (_Float16)p0;
      Pw[(r + 8 * half) * 32 + col + 16] = (_Float16)p1;
    }
    // ---- reload P as a 16x32 A-fragment from LDS
    v16h pf;
    {
      const _Float16* qp = Pw + (lane & 15) * 32 + ((lane >> 4) << 3);
#pragma unroll
      for (int i = 0; i < 8; ++i) { pf[i] = qp[i]; pf[i + 8] = qp[i + 16]; }
    }
    // ---- O += P @ V (V transposed: rows = d, contiguous over keys)
#pragma unroll
    for (int t = 0; t < 4; ++t)
      o[t] = wmma16(pf, load_frag(Vp + (size_t)(16 * t) * S_ + k0, S_, lane), o[t]);
  }

  // ---- normalize, store O as [b][s][h*64+d] for the output GEMM
#pragma unroll
  for (int t = 0; t < 4; ++t)
#pragma unroll
    for (int r = 0; r < 8; ++r) {
      const float val = o[t][r] / fmaxf(rl[r], 1e-20f);
      const int srow = q0 + r + 8 * half;
      O16[((size_t)bb * S_ + srow) * DIM_ + hh * HD_ + 16 * t + col] = (_Float16)val;
    }
}

// ------------------------------------------------- Out = O @ Wo^T (f32 out)
// grid (BS_/128, DIM_/128), same wave tiling as qkv_gemm.
__global__ __launch_bounds__(256) void out_gemm(const _Float16* __restrict__ O16,
                                                const _Float16* __restrict__ wo16,
                                                float* __restrict__ out) {
  const int lane = threadIdx.x & 31, wave = threadIdx.x >> 5;
  const int m0 = blockIdx.x * 128 + (wave & 1) * 64;
  const int n0 = blockIdx.y * 128 + (wave >> 1) * 32;
  v8f acc[4][2] = {};
  for (int k0 = 0; k0 < DIM_; k0 += 32) {
    __builtin_prefetch(O16 + (size_t)m0 * DIM_ + k0 + 64, 0, 1);
    __builtin_prefetch(wo16 + (size_t)n0 * DIM_ + k0 + 64, 0, 1);
    v16h a[4], b[2];
#pragma unroll
    for (int im = 0; im < 4; ++im)
      a[im] = load_frag(O16 + (size_t)(m0 + 16 * im) * DIM_ + k0, DIM_, lane);
#pragma unroll
    for (int in = 0; in < 2; ++in)
      b[in] = load_frag(wo16 + (size_t)(n0 + 16 * in) * DIM_ + k0, DIM_, lane);
#pragma unroll
    for (int im = 0; im < 4; ++im)
#pragma unroll
      for (int in = 0; in < 2; ++in)
        acc[im][in] = wmma16(a[im], b[in], acc[im][in]);
  }
  const int half = lane >> 4, col = lane & 15;
#pragma unroll
  for (int im = 0; im < 4; ++im)
#pragma unroll
    for (int in = 0; in < 2; ++in)
#pragma unroll
      for (int r = 0; r < 8; ++r) {
        const int m = m0 + 16 * im + r + 8 * half;
        const int e = n0 + 16 * in + col;
        out[(size_t)m * DIM_ + e] = acc[im][in][r];
      }
}

extern "C" void kernel_launch(void* const* d_in, const int* in_sizes, int n_in,
                              void* d_out, int out_size, void* d_ws, size_t ws_size,
                              hipStream_t stream) {
  const float* x            = (const float*)d_in[0];
  const unsigned char* mask = (const unsigned char*)d_in[1];  // jnp bool = 1 byte
  const float* wqkv         = (const float*)d_in[2];
  const float* wo           = (const float*)d_in[3];
  float* out                = (float*)d_out;

  // workspace carving (f16, ~88 MB total)
  _Float16* x16  = (_Float16*)d_ws;
  _Float16* wq16 = x16  + (size_t)BS_ * DIM_;          // 8,388,608
  _Float16* wo16 = wq16 + (size_t)E3_ * DIM_;          // 3,145,728
  _Float16* Qf   = wo16 + (size_t)DIM_ * DIM_;         // 1,048,576
  _Float16* Kf   = Qf   + (size_t)B_ * NH_ * S_ * HD_;
  _Float16* Vt   = Kf   + (size_t)B_ * NH_ * S_ * HD_;
  _Float16* O16  = Vt   + (size_t)B_ * NH_ * S_ * HD_;

  const int nx = BS_ * DIM_, nwq = E3_ * DIM_, nwo = DIM_ * DIM_;
  cvt_f32_f16<<<(nx / 4 + 255) / 256, 256, 0, stream>>>(x, x16, nx);
  cvt_f32_f16<<<(nwq / 4 + 255) / 256, 256, 0, stream>>>(wqkv, wq16, nwq);
  cvt_f32_f16<<<(nwo / 4 + 255) / 256, 256, 0, stream>>>(wo, wo16, nwo);

  qkv_gemm<<<dim3(BS_ / 128, E3_ / 128), 256, 0, stream>>>(x16, wq16, Qf, Kf, Vt);
  attn_kernel<<<dim3(S_ / 128, B_ * NH_), 256, 0, stream>>>(Qf, Kf, Vt, mask, O16);
  out_gemm<<<dim3(BS_ / 128, DIM_ / 128), 256, 0, stream>>>(O16, wo16, out);
}
